// GNNConv_13297218748565
// MI455X (gfx1250) — matrix operand order repaired and checked by reading, not compiled, not run on hardware
//
#include <hip/hip_runtime.h>

typedef __attribute__((ext_vector_type(16))) __bf16 v16bf;
typedef __attribute__((ext_vector_type(8)))  __bf16 v8bf;
typedef __attribute__((ext_vector_type(4)))  __bf16 v4bf;
typedef __attribute__((ext_vector_type(8)))  float  v8f;
typedef __attribute__((ext_vector_type(4)))  float  v4f;

#define EMB     128
#define NNODES  50000
#define NEDGES  800000
#define NW      4        // waves per block (wave32)
#define MT      2        // M-tiles per wave (B-fragment reuse factor)
#define TS      16       // rows per M-tile
#define MROWS   (MT*TS)  // 32 rows per wave
#define LDSS    136      // LDS row stride in bf16 elems (16B-aligned rows, padded)

struct AFrag { v16bf k[4]; };

// Build one 16x32 bf16 A-fragment from an LDS row (contiguous K order, same as B in wmma_ntile2):
// lanes 0-15 : K = k0+0..15
// lanes 16-31: K = k0+16..31
__device__ __forceinline__ v16bf make_a(const __bf16* row, int k0, int hi) {
  v8bf lo = *(const v8bf*)(row + k0 + hi * 16);
  v8bf hc = *(const v8bf*)(row + k0 + hi * 16 + 8);
  v16bf a;
#pragma unroll
  for (int i = 0; i < 8; ++i) { a[i] = lo[i]; a[i + 8] = hc[i]; }
  return a;
}

// One N-tile for MT=2 M-tiles sharing every B fragment (1 b128-pair per 2 WMMAs).
// B[k][n] = W[n][k]: per-lane 16 contiguous bf16 from row (n) of W.
__device__ __forceinline__ void wmma_ntile2(const AFrag& a0, const AFrag& a1,
                                            const __bf16* __restrict__ W,
                                            int n, int hi, v8f& acc0, v8f& acc1) {
#pragma unroll
  for (int kk = 0; kk < 4; ++kk) {
    v16bf b = *(const v16bf*)(W + n * EMB + kk * 32 + hi * 16);
    acc0 = __builtin_amdgcn_wmma_f32_16x16x32_bf16(false, a0.k[kk], false, b,
                                                   (short)0, acc0, false, false);
    acc1 = __builtin_amdgcn_wmma_f32_16x16x32_bf16(false, a1.k[kk], false, b,
                                                   (short)0, acc1, false, false);
  }
}

// ---------------------------------------------------------------------------
// Prep: zero the aggregation surface (d_out doubles as agg) + f32->bf16 weights
// ---------------------------------------------------------------------------
__global__ void gnn_prep_kernel(const float* __restrict__ W1f, const float* __restrict__ W2f,
                                const float* __restrict__ W3f, const float* __restrict__ W4f,
                                __bf16* __restrict__ W1b, __bf16* __restrict__ W2b,
                                __bf16* __restrict__ W3b, __bf16* __restrict__ W4b,
                                float* __restrict__ agg) {
  const long i = (long)blockIdx.x * blockDim.x + threadIdx.x;
  if (i < (long)NNODES * EMB) agg[i] = 0.f;
  if (i < EMB * EMB) {
    W1b[i] = (__bf16)W1f[i];
    W2b[i] = (__bf16)W2f[i];
    W3b[i] = (__bf16)W3f[i];
    W4b[i] = (__bf16)W4f[i];
  }
}

// ---------------------------------------------------------------------------
// Fused edge pipeline: he -> relu(he@W1^T+b1)@W2^T+b2 -> *h[src] -> atomic agg[dst]
// Each wave owns 32 edges (2 M-tiles); X and Y share one LDS buffer (A tile
// lives entirely in registers once fragments are built; same-wave DS is in-order).
// ---------------------------------------------------------------------------
__launch_bounds__(NW * 32)
__global__ void gnn_edge_kernel(const float* __restrict__ h,
                                const float* __restrict__ he,
                                const int*   __restrict__ src,
                                const int*   __restrict__ dst,
                                const __bf16* __restrict__ W1,
                                const float* __restrict__ b1,
                                const __bf16* __restrict__ W2,
                                const float* __restrict__ b2,
                                float* __restrict__ agg) {
  __shared__ __bf16 Ts[NW][MROWS][LDSS];
  __shared__ int SrcS[NW][MROWS];
  __shared__ int DstS[NW][MROWS];

  const int lane = threadIdx.x & 31;
  const int wave = threadIdx.x >> 5;
  const int hl   = lane & 15;
  const int hi   = lane >> 4;
  const long e0  = ((long)blockIdx.x * NW + wave) * MROWS;

  SrcS[wave][lane] = src[e0 + lane];
  DstS[wave][lane] = dst[e0 + lane];

  // Warm L2 with layer-2 weights while layer 1 runs (global_prefetch_b8).
  __builtin_prefetch(W2 + (size_t)threadIdx.x * 128, 0, 1);

  // Stage 32x128 he tile into LDS as bf16 (coalesced: 32 lanes * 16B per row).
#pragma unroll
  for (int r = 0; r < MROWS; ++r) {
    v4f x = *(const v4f*)(he + (e0 + r) * EMB + lane * 4);
    v4bf xb;
#pragma unroll
    for (int i = 0; i < 4; ++i) xb[i] = (__bf16)x[i];
    *(v4bf*)&Ts[wave][r][lane * 4] = xb;
  }

  AFrag a[MT];
#pragma unroll
  for (int m = 0; m < MT; ++m)
#pragma unroll
    for (int kk = 0; kk < 4; ++kk)
      a[m].k[kk] = make_a(&Ts[wave][m * TS + hl][0], kk * 32, hi);

  // Layer 1: relu(X@W1^T + b1) -> Y (bf16), reusing the same LDS buffer
#pragma unroll
  for (int nt = 0; nt < 8; ++nt) {
    const int n = nt * 16 + hl;
    v8f acc0 = {0.f,0.f,0.f,0.f,0.f,0.f,0.f,0.f};
    v8f acc1 = {0.f,0.f,0.f,0.f,0.f,0.f,0.f,0.f};
    wmma_ntile2(a[0], a[1], W1, n, hi, acc0, acc1);
    const float bb = b1[n];
#pragma unroll
    for (int r = 0; r < 8; ++r) {       // C/D layout: row = r + 8*hi, col = n
      float y0 = acc0[r] + bb; y0 = y0 > 0.f ? y0 : 0.f;
      float y1 = acc1[r] + bb; y1 = y1 > 0.f ? y1 : 0.f;
      Ts[wave][r + hi * 8][n]      = (__bf16)y0;
      Ts[wave][TS + r + hi * 8][n] = (__bf16)y1;
    }
  }

  AFrag c[MT];
#pragma unroll
  for (int m = 0; m < MT; ++m)
#pragma unroll
    for (int kk = 0; kk < 4; ++kk)
      c[m].k[kk] = make_a(&Ts[wave][m * TS + hl][0], kk * 32, hi);

  // Layer 2 + gather-multiply + scatter-add (agg fits in L2: atomics stay on-chip)
#pragma unroll
  for (int nt = 0; nt < 8; ++nt) {
    const int n = nt * 16 + hl;
    v8f acc0 = {0.f,0.f,0.f,0.f,0.f,0.f,0.f,0.f};
    v8f acc1 = {0.f,0.f,0.f,0.f,0.f,0.f,0.f,0.f};
    wmma_ntile2(c[0], c[1], W2, n, hi, acc0, acc1);
    const float bb = b2[n];
#pragma unroll
    for (int m = 0; m < MT; ++m) {
#pragma unroll
      for (int r = 0; r < 8; ++r) {
        const int row = m * TS + r + hi * 8;
        const int s = SrcS[wave][row];
        const int d = DstS[wave][row];
        const float z = (m == 0 ? acc0[r] : acc1[r]) + bb;
        const float msg = z * h[(long)s * EMB + n];
        atomicAdd(&agg[(long)d * EMB + n], msg);
      }
    }
  }
}

// ---------------------------------------------------------------------------
// Node MLP: out = relu(agg@W3^T+b3)@W4^T + b4  (in-place over agg rows)
// ---------------------------------------------------------------------------
__launch_bounds__(NW * 32)
__global__ void gnn_node_kernel(const float* __restrict__ agg,
                                const __bf16* __restrict__ W3,
                                const float* __restrict__ b3,
                                const __bf16* __restrict__ W4,
                                const float* __restrict__ b4,
                                float* __restrict__ out) {
  __shared__ __bf16 Ts[NW][MROWS][LDSS];

  const int lane = threadIdx.x & 31;
  const int wave = threadIdx.x >> 5;
  const int hl   = lane & 15;
  const int hi   = lane >> 4;
  const long row0 = ((long)blockIdx.x * NW + wave) * MROWS;

  __builtin_prefetch(W4 + (size_t)threadIdx.x * 128, 0, 1);

#pragma unroll
  for (int r = 0; r < MROWS; ++r) {
    const long row = row0 + r;
    v4f x = {0.f, 0.f, 0.f, 0.f};
    if (row < NNODES) x = *(const v4f*)(agg + row * EMB + lane * 4);
    v4bf xb;
#pragma unroll
    for (int i = 0; i < 4; ++i) xb[i] = (__bf16)x[i];
    *(v4bf*)&Ts[wave][r][lane * 4] = xb;
  }

  AFrag a[MT];
#pragma unroll
  for (int m = 0; m < MT; ++m)
#pragma unroll
    for (int kk = 0; kk < 4; ++kk)
      a[m].k[kk] = make_a(&Ts[wave][m * TS + hl][0], kk * 32, hi);

#pragma unroll
  for (int nt = 0; nt < 8; ++nt) {
    const int n = nt * 16 + hl;
    v8f acc0 = {0.f,0.f,0.f,0.f,0.f,0.f,0.f,0.f};
    v8f acc1 = {0.f,0.f,0.f,0.f,0.f,0.f,0.f,0.f};
    wmma_ntile2(a[0], a[1], W3, n, hi, acc0, acc1);
    const float bb = b3[n];
#pragma unroll
    for (int r = 0; r < 8; ++r) {
      float y0 = acc0[r] + bb; y0 = y0 > 0.f ? y0 : 0.f;
      float y1 = acc1[r] + bb; y1 = y1 > 0.f ? y1 : 0.f;
      Ts[wave][r + hi * 8][n]      = (__bf16)y0;
      Ts[wave][TS + r + hi * 8][n] = (__bf16)y1;
    }
  }

  AFrag c[MT];
#pragma unroll
  for (int m = 0; m < MT; ++m)
#pragma unroll
    for (int kk = 0; kk < 4; ++kk)
      c[m].k[kk] = make_a(&Ts[wave][m * TS + hl][0], kk * 32, hi);

#pragma unroll
  for (int nt = 0; nt < 8; ++nt) {
    const int n = nt * 16 + hl;
    v8f acc0 = {0.f,0.f,0.f,0.f,0.f,0.f,0.f,0.f};
    v8f acc1 = {0.f,0.f,0.f,0.f,0.f,0.f,0.f,0.f};
    wmma_ntile2(c[0], c[1], W4, n, hi, acc0, acc1);
    const float bb = b4[n];
#pragma unroll
    for (int m = 0; m < MT; ++m) {
#pragma unroll
      for (int r = 0; r < 8; ++r) {
        const long row = row0 + m * TS + r + hi * 8;
        const float z = (m == 0 ? acc0[r] : acc1[r]) + bb;
        if (row < NNODES) out[row * EMB + n] = z;
      }
    }
  }
}

// ---------------------------------------------------------------------------
extern "C" void kernel_launch(void* const* d_in, const int* in_sizes, int n_in,
                              void* d_out, int out_size, void* d_ws, size_t ws_size,
                              hipStream_t stream) {
  (void)in_sizes; (void)n_in; (void)out_size; (void)ws_size;

  const float* h   = (const float*)d_in[0];
  const float* he  = (const float*)d_in[1];
  const int*   src = (const int*)d_in[2];
  const int*   dst = (const int*)d_in[3];
  const float* W1  = (const float*)d_in[4];
  const float* b1  = (const float*)d_in[5];
  const float* W2  = (const float*)d_in[6];
  const float* b2  = (const float*)d_in[7];
  const float* W3  = (const float*)d_in[8];
  const float* b3  = (const float*)d_in[9];
  const float* W4  = (const float*)d_in[10];
  const float* b4  = (const float*)d_in[11];

  float* out = (float*)d_out;       // also used as the agg accumulator (25.6 MB)

  char* ws = (char*)d_ws;           // bf16 weights: 4 * 128*128*2 = 128 KB scratch
  __bf16* W1b = (__bf16*)(ws);
  __bf16* W2b = (__bf16*)(ws + 1 * EMB * EMB * sizeof(__bf16));
  __bf16* W3b = (__bf16*)(ws + 2 * EMB * EMB * sizeof(__bf16));
  __bf16* W4b = (__bf16*)(ws + 3 * EMB * EMB * sizeof(__bf16));

  // 1) zero agg(out), convert weights to bf16
  {
    const long total = (long)NNODES * EMB;              // 6.4M
    const int blocks = (int)((total + 255) / 256);
    gnn_prep_kernel<<<blocks, 256, 0, stream>>>(W1, W2, W3, W4,
                                                W1b, W2b, W3b, W4b, out);
  }
  // 2) fused edge MLP + gather-multiply + scatter-sum (32 edges per wave)
  gnn_edge_kernel<<<NEDGES / (NW * MROWS), NW * 32, 0, stream>>>(
      h, he, src, dst, W1b, b1, W2b, b2, out);
  // 3) node MLP (in-place over agg rows)
  gnn_node_kernel<<<(NNODES + NW * MROWS - 1) / (NW * MROWS), NW * 32, 0, stream>>>(
      out, W3b, b3, W4b, b4, out);
}
